// MultiLayer_LSTM_45286135169414
// MI455X (gfx1250) — matrix operand order repaired and checked by reading, not compiled, not run on hardware
//
#include <hip/hip_runtime.h>

// ---------------------------------------------------------------------------
// MultiLayer LSTM (2 blocks x [LSTM -> LN -> LSTM -> LN -> Highway]) for
// MI455X / gfx1250.  All matmuls via v_wmma_f32_16x16x32_bf16 (wave32).
// Recurrence keeps U weight slices resident in LDS (320KB/WGP) in pre-swizzled
// WMMA B-fragment layout for all 512 timesteps; h(t) is exchanged between the
// 16 cooperating workgroups of a batch-group through L2 in A-fragment layout,
// staged into LDS each step by the Tensor Data Mover (tensor_load_to_lds).
// ---------------------------------------------------------------------------

typedef __attribute__((ext_vector_type(16))) __bf16 v16bf;
typedef __attribute__((ext_vector_type(8)))  float  v8f;
typedef unsigned int u32x4 __attribute__((ext_vector_type(4)));
typedef int          i32x4 __attribute__((ext_vector_type(4)));
typedef int          i32x8 __attribute__((ext_vector_type(8)));

#define DEVINL __device__ __forceinline__

constexpr int kB  = 64;
constexpr int kT  = 512;
constexpr int kF  = 128;
constexpr int kU  = 512;
constexpr int k4U = 2048;
constexpr int kBT = kB * kT;          // 32768 rows
constexpr int kOutCols = kF + 2 * kU; // 1152
constexpr unsigned kRecLds = 157696;  // 128K U-frags + 16K h-frags + 8K z + 2K c

// Fast activations on the serial critical path: v_exp_f32 + v_rcp_f32 instead
// of the IEEE division expansion (v_div_scale/v_div_fmas chain) and ocml tanh.
DEVINL float fast_rcp(float x) {
#if __has_builtin(__builtin_amdgcn_rcpf)
  return __builtin_amdgcn_rcpf(x);
#else
  return __fdividef(1.0f, x);
#endif
}
DEVINL float sigf(float x) { return fast_rcp(1.0f + __expf(-x)); }
DEVINL float tanh_fast(float x) {
  // tanh(x) = 2*sigmoid(2x) - 1
  float t = fast_rcp(1.0f + __expf(-2.0f * x));
  return __builtin_fmaf(2.0f, t, -1.0f);
}

DEVINL v8f wmma_bf16(v16bf a, v16bf b, v8f c) {
  // D = A(16x32) * B(32x16) + C(16x16 f32)
  return __builtin_amdgcn_wmma_f32_16x16x32_bf16(false, a, false, b,
                                                 (short)0, c, false, false);
}

// Monotonic cross-workgroup barrier (16 members per batch-group).  The
// s_cluster_barrier documents the intended wave-cluster sync path; per ISA it
// is an S_NOP when ClusterID==0, so it is safe for non-cluster dispatch while
// the agent-scope atomic provides the guaranteed ordering.
DEVINL void bgroup_barrier(unsigned* p, unsigned target) {
  __syncthreads();
  if (threadIdx.x == 0) {
    __hip_atomic_fetch_add(p, 1u, __ATOMIC_ACQ_REL, __HIP_MEMORY_SCOPE_AGENT);
    while (__hip_atomic_load(p, __ATOMIC_ACQUIRE, __HIP_MEMORY_SCOPE_AGENT) < target)
      __builtin_amdgcn_s_sleep(2);
  }
#if __has_builtin(__builtin_amdgcn_s_cluster_barrier)
  __builtin_amdgcn_s_cluster_barrier();
#endif
  __syncthreads();
}

// ---- Tensor Data Mover: 16KB contiguous global->LDS copy as a 1-D tensor
// tile (2048 x 8B).  Arity differs across toolchains; the TDM header ships
// only with the 6-arg toolchain, so probe for it.
#if __has_builtin(__builtin_amdgcn_tensor_load_to_lds)
#define LSTM_HAVE_TDM 1
DEVINL void tdm_copy_16k(const void* gsrc, unsigned lds_byte_addr) {
  unsigned long long ga = (unsigned long long)(__SIZE_TYPE__)gsrc;
  u32x4 g0;
  g0[0] = 1u;                                  // count=1 user descriptor
  g0[1] = lds_byte_addr;                       // lds_addr
  g0[2] = (unsigned)(ga & 0xFFFFFFFFu);        // global_addr[31:0]
  g0[3] = (unsigned)((ga >> 32) & 0x1FFFFFFu)  // global_addr[56:32]
          | (2u << 30);                        // type=2 ("image")
  i32x8 g1;
  g1[0] = 3 << 16;                             // data_size = 8 bytes
  g1[1] = (int)(2048u << 16);                  // tensor_dim0[15:0] @ bits 63:48
  g1[2] = (int)(1u << 16);                     // tensor_dim0 hi=0; tensor_dim1=1
  g1[3] = (int)(2048u << 16);                  // tile_dim0 = 2048
  g1[4] = 0;                                   // tile_dim1/2 unused
  g1[5] = 2048;                                // tensor_dim0_stride
  g1[6] = 0;
  g1[7] = 0;
  i32x4 z4 = {0, 0, 0, 0};
#if __has_include(<hip/amd_detail/amd_gfx1250_TDM.h>)
  i32x8 z8 = {0, 0, 0, 0, 0, 0, 0, 0};
  __builtin_amdgcn_tensor_load_to_lds(g0, g1, z4, z4, z8, 0);
#else
  __builtin_amdgcn_tensor_load_to_lds(g0, g1, z4, z4, 0);
#endif
}
DEVINL void tdm_wait() {
#if __has_builtin(__builtin_amdgcn_s_wait_tensorcnt)
  __builtin_amdgcn_s_wait_tensorcnt(0);
#else
  asm volatile("s_wait_tensorcnt 0x0" ::: "memory");
#endif
}
#endif

// ---------------------------------------------------------------------------
// fp32 -> bf16 conversion (weights / activations)
// ---------------------------------------------------------------------------
__global__ void convert_bf16_kernel(const float* __restrict__ src,
                                    __bf16* __restrict__ dst, size_t n) {
  size_t stride = (size_t)gridDim.x * blockDim.x;
  for (size_t i = (size_t)blockIdx.x * blockDim.x + threadIdx.x; i < n; i += stride)
    dst[i] = (__bf16)src[i];
}

// x -> d_out[:, 0:128], x_bf16, in1_bf16[:, 0:128]
__global__ void scatter_x_kernel(const float* __restrict__ x, float* __restrict__ dOut,
                                 __bf16* __restrict__ xbf, __bf16* __restrict__ in1bf) {
  size_t stride = (size_t)gridDim.x * blockDim.x;
  for (size_t i = (size_t)blockIdx.x * blockDim.x + threadIdx.x;
       i < (size_t)kBT * kF; i += stride) {
    size_t row = i / kF;
    int col = (int)(i % kF);
    float v = x[i];
    dOut[row * kOutCols + col] = v;
    xbf[i] = (__bf16)v;
    in1bf[row * 640 + col] = (__bf16)v;
  }
}

// ---------------------------------------------------------------------------
// Generic bf16 WMMA GEMM:  C[M,N] = A[M,K] @ B[K,N] + bias  (opt. sigmoid)
// block tile 64x128, 8 waves, each wave -> 4 (16x16) accumulators.
// LDS staging writes A/B directly in WMMA fragment order.
// ---------------------------------------------------------------------------
__global__ __launch_bounds__(256) void gemm_bf16_kernel(
    const __bf16* __restrict__ A, const __bf16* __restrict__ Bw,
    const float* __restrict__ bias, float* __restrict__ C,
    int M, int N, int K, int act) {
  __shared__ __align__(32) char sA[4096];  // 4 m-tiles  x (32 lanes x 32B)
  __shared__ __align__(32) char sB[8192];  // 8 n-tiles  x (32 lanes x 32B)
  const int m0 = blockIdx.y * 64;
  const int n0 = blockIdx.x * 128;
  const int tid = threadIdx.x;
  const int w = tid >> 5, lane = tid & 31;
  const int mt = w & 3, ng = w >> 2;

  v8f acc[4] = {};
  for (int k0 = 0; k0 < K; k0 += 32) {
    __syncthreads();
    { // ---- stage A tile (64x32 bf16) in A-fragment order: 16B per thread
      int row = tid >> 2, s = tid & 3;  // s -> k run [8s, 8s+8)
      uint4 v = *(const uint4*)(A + (size_t)(m0 + row) * K + k0 + s * 8);
      int lanei = ((s & 1) << 4) | (row & 15);          // k%16>=8 -> hi half-wave
      *(uint4*)(sA + ((row >> 4) << 10) + lanei * 32 + ((s >> 1) << 4)) = v;
    }
    { // ---- stage B tile (32x128 bf16) in B-fragment order
      int kk = tid >> 3, s = tid & 7;  // row k = kk, cols [16s, 16s+16)
      const __bf16* src = Bw + (size_t)(k0 + kk) * N + n0 + s * 16;
      __bf16 tmp[16] __attribute__((aligned(16)));
      *(uint4*)(tmp)     = *(const uint4*)(src);
      *(uint4*)(tmp + 8) = *(const uint4*)(src + 8);
      int laneBase = ((kk >> 3) & 1) << 4;   // k%16>=8 -> lanes 16..31
      int e = ((kk >> 4) << 3) | (kk & 7);
      char* dst = sB + (s << 10) + laneBase * 32 + e * 2;
#pragma unroll
      for (int j = 0; j < 16; ++j) *(__bf16*)(dst + j * 32) = tmp[j];
    }
    __syncthreads();
    v16bf a = *(const v16bf*)(sA + (mt << 10) + lane * 32);
#pragma unroll
    for (int q = 0; q < 4; ++q) {
      v16bf bb = *(const v16bf*)(sB + ((ng * 4 + q) << 10) + lane * 32);
      acc[q] = wmma_bf16(a, bb, acc[q]);
    }
  }
  // ---- epilogue: C-fragment layout -> global, fused bias (+sigmoid)
  int ncol = lane & 15;
  int mrow = m0 + mt * 16 + ((lane >> 4) << 3);
#pragma unroll
  for (int q = 0; q < 4; ++q) {
    int n = n0 + (ng * 4 + q) * 16 + ncol;
    float bs = bias[n];
#pragma unroll
    for (int r = 0; r < 8; ++r) {
      float v = acc[q][r] + bs;
      if (act) v = sigf(v);
      C[(size_t)(mrow + r) * N + n] = v;
    }
  }
}

// ---------------------------------------------------------------------------
// Recurrence init: zero h(0) fragment buffers + barrier counters
// ---------------------------------------------------------------------------
__global__ void init_rec_kernel(__bf16* __restrict__ hfrag, unsigned* __restrict__ ctr) {
  int i = blockIdx.x * blockDim.x + threadIdx.x;
  int stride = gridDim.x * blockDim.x;
  unsigned short* h = (unsigned short*)hfrag;
  for (int j = i; j < 4 * 8192; j += stride) h[j] = 0;
  if (i < 4) ctr[i] = 0u;
}

// ---------------------------------------------------------------------------
// LSTM recurrence.  Grid = 64 WGs: bg = blockIdx&3 (16 batch rows),
// jg = blockIdx>>2 (32 hidden cols -> 128 z cols across the 4 gates).
// U slice (512x128) lives in LDS as B-fragments for all 512 steps.
// ---------------------------------------------------------------------------
__global__ __launch_bounds__(256) void lstm_rec_kernel(
    const float* __restrict__ xW, const float* __restrict__ Ug,
    float* __restrict__ h_seq, __bf16* __restrict__ hfrag,
    unsigned* __restrict__ ctr) {
  extern __shared__ char smem[];
  char*  sU = smem;                      // 128KB: [kt(16)][nt(8)] x 1KB B-frag tiles
  char*  sH = smem + 131072;             // 16KB : A-frags of h(t), 16 k-tiles
  float* sZ = (float*)(smem + 147456);   // 16x128 pre-activations
  float* sC = (float*)(smem + 155648);   // 16x32 cell state

  const int tid = threadIdx.x;
  const int bg = blockIdx.x & 3;
  const int jg = blockIdx.x >> 2;
  const int j0 = jg * 32;

  // ---- one-time: load + bf16-convert + swizzle U slice into B-fragment LDS
  for (int idx = tid; idx < kU * 128; idx += 256) {
    int k = idx >> 7;
    int n = idx & 127;                   // local z col: gate*32 + c
    int gate = n >> 5, c = n & 31;
    float v = Ug[(size_t)k * k4U + gate * kU + j0 + c];
    int kt = k >> 5, kl = k & 31;
    int lane = (((kl >> 3) & 1) << 4) | (n & 15);
    int e = ((kl >> 4) << 3) | (kl & 7);
    *(__bf16*)(sU + ((kt * 8 + (n >> 4)) << 10) + lane * 32 + e * 2) = (__bf16)v;
  }
  for (int i = tid; i < 16 * 32; i += 256) sC[i] = 0.0f;
  __syncthreads();

  const int w = tid >> 5, lane = tid & 31;
  const int gate = w >> 1;
  unsigned* bar = ctr + bg;

  for (int t = 0; t < kT; ++t) {
    // h(t) published by all 16 producers of this batch-group
    bgroup_barrier(bar, 32u * (unsigned)t + 16u);
    // ---- stage h(t): 16KB copy, already in A-fragment order
#if defined(LSTM_HAVE_TDM)
    if (w == 0) {
      tdm_copy_16k(hfrag + bg * 8192, (unsigned)(__SIZE_TYPE__)sH);
      tdm_wait();
    }
#else
    {
      const uint4* src = (const uint4*)(hfrag + bg * 8192);
      uint4* dst = (uint4*)sH;
      for (int i = tid; i < 1024; i += 256) dst[i] = src[i];
    }
#endif
    __syncthreads();
    bgroup_barrier(bar, 32u * (unsigned)t + 32u);  // h(t) consumed everywhere

    // ---- z_tile = h(t) @ U_slice : K=512 -> 16 chained WMMA
    v8f acc = {};
#pragma unroll
    for (int kt = 0; kt < 16; ++kt) {
      v16bf a = *(const v16bf*)(sH + (kt << 10) + lane * 32);
      v16bf b = *(const v16bf*)(sU + ((kt * 8 + w) << 10) + lane * 32);
      acc = wmma_bf16(a, b, acc);
    }

    // ---- add input projection, park pre-activations in LDS
    {
      int nloc = (w << 4) | (lane & 15);
      int c = nloc - gate * 32;
      int gcol = gate * kU + j0 + c;
      int mbase = (lane >> 4) << 3;
#pragma unroll
      for (int r = 0; r < 8; ++r) {
        int m = mbase + r;
        size_t row = (size_t)(bg * 16 + m) * kT + t;
        sZ[m * 128 + nloc] = acc[r] + xW[row * k4U + gcol];
      }
      if (t + 1 < kT)  // global_prefetch_b8 of next step's xW line
        __builtin_prefetch(xW + ((size_t)(bg * 16 + mbase) * kT + t + 1) * k4U + gcol, 0, 1);
    }
    __syncthreads();

    // ---- gate math + state update: thread -> column c, rows {mh, mh+8}
    {
      int c = tid & 31;
      int mh = tid >> 5;
#pragma unroll
      for (int p = 0; p < 2; ++p) {
        int m = mh + (p << 3);
        float zi = sZ[m * 128 + c];
        float zf = sZ[m * 128 + 32 + c];
        float zg = sZ[m * 128 + 64 + c];
        float zo = sZ[m * 128 + 96 + c];
        float cs = sigf(zf) * sC[m * 32 + c] + sigf(zi) * tanh_fast(zg);
        sC[m * 32 + c] = cs;
        float h = sigf(zo) * tanh_fast(cs);
        size_t row = (size_t)(bg * 16 + m) * kT + t;
        h_seq[row * kU + j0 + c] = h;
        // scatter h(t+1) into A-fragment layout for the next step
        int k = j0 + c;
        int kt2 = k >> 5, kl = k & 31;
        int lane2 = (((kl & 15) >= 8) ? 16 : 0) + m;
        int e2 = ((kl >> 4) << 3) | (kl & 7);
        hfrag[bg * 8192 + (kt2 << 9) + lane2 * 16 + e2] = (__bf16)h;
      }
    }
    __threadfence();
    __syncthreads();
  }
}

// ---------------------------------------------------------------------------
// LayerNorm over last dim (512); writes fp32 + optional bf16 copy
// ---------------------------------------------------------------------------
__global__ __launch_bounds__(256) void layernorm_kernel(
    const float* __restrict__ in, const float* __restrict__ g,
    const float* __restrict__ be, float* __restrict__ outf,
    __bf16* __restrict__ outb) {
  __shared__ float s1[256];
  __shared__ float s2[256];
  const size_t row = blockIdx.x;
  const int tid = threadIdx.x;
  float a = in[row * kU + tid];
  float b = in[row * kU + 256 + tid];
  s1[tid] = a + b;
  s2[tid] = a * a + b * b;
  __syncthreads();
  for (int s = 128; s > 0; s >>= 1) {
    if (tid < s) { s1[tid] += s1[tid + s]; s2[tid] += s2[tid + s]; }
    __syncthreads();
  }
  float mean = s1[0] * (1.0f / kU);
  float var  = s2[0] * (1.0f / kU) - mean * mean;
  float inv  = rsqrtf(var + 1e-3f);
  float o1 = (a - mean) * inv * g[tid] + be[tid];
  float o2 = (b - mean) * inv * g[256 + tid] + be[256 + tid];
  outf[row * kU + tid] = o1;
  outf[row * kU + 256 + tid] = o2;
  if (outb) {
    outb[row * kU + tid] = (__bf16)o1;
    outb[row * kU + 256 + tid] = (__bf16)o2;
  }
}

// ---------------------------------------------------------------------------
// Highway merge: hw = Hx*Tx + (1-Tx)*out  -> d_out slice (+ bf16 for next blk)
// ---------------------------------------------------------------------------
__global__ void highway_kernel(const float* __restrict__ Hx, const float* __restrict__ Tx,
                               const float* __restrict__ out, float* __restrict__ dOut,
                               int colOff, __bf16* __restrict__ nextbf) {
  size_t stride = (size_t)gridDim.x * blockDim.x;
  for (size_t i = (size_t)blockIdx.x * blockDim.x + threadIdx.x;
       i < (size_t)kBT * kU; i += stride) {
    size_t row = i / kU;
    int col = (int)(i % kU);
    float tg = Tx[i];
    float hw = Hx[i] * tg + (1.0f - tg) * out[i];
    dOut[row * kOutCols + colOff + col] = hw;
    if (nextbf) nextbf[row * 640 + 128 + col] = (__bf16)hw;
  }
}

// ---------------------------------------------------------------------------
// Host driver
// ---------------------------------------------------------------------------
extern "C" void kernel_launch(void* const* d_in, const int* in_sizes, int n_in,
                              void* d_out, int out_size, void* d_ws, size_t ws_size,
                              hipStream_t stream) {
  (void)in_sizes; (void)n_in; (void)out_size; (void)ws_size;
  const float* x   = (const float*)d_in[0];
  const float* W_[4]  = {(const float*)d_in[1],  (const float*)d_in[6],
                         (const float*)d_in[11], (const float*)d_in[16]};
  const float* U_[4]  = {(const float*)d_in[2],  (const float*)d_in[7],
                         (const float*)d_in[12], (const float*)d_in[17]};
  const float* b_[4]  = {(const float*)d_in[3],  (const float*)d_in[8],
                         (const float*)d_in[13], (const float*)d_in[18]};
  const float* g_[4]  = {(const float*)d_in[4],  (const float*)d_in[9],
                         (const float*)d_in[14], (const float*)d_in[19]};
  const float* be_[4] = {(const float*)d_in[5],  (const float*)d_in[10],
                         (const float*)d_in[15], (const float*)d_in[20]};
  const float* Wt_[2] = {(const float*)d_in[21], (const float*)d_in[23]};
  const float* bt_[2] = {(const float*)d_in[22], (const float*)d_in[24]};
  float* out = (float*)d_out;

  // ---- workspace carve-out
  char* ws = (char*)d_ws;
  size_t off = 0;
  auto carve = [&](size_t bytes) -> char* {
    char* p = ws + off;
    off += (bytes + 255) & ~(size_t)255;
    return p;
  };
  float*  xW    = (float*)carve((size_t)kBT * k4U * 4);
  float*  hseq  = (float*)carve((size_t)kBT * kU * 4);
  float*  lnA   = (float*)carve((size_t)kBT * kU * 4);
  float*  lnB   = (float*)carve((size_t)kBT * kU * 4);
  float*  TxB   = (float*)carve((size_t)kBT * kU * 4);
  __bf16* lnAb  = (__bf16*)carve((size_t)kBT * kU * 2);
  __bf16* xbf   = (__bf16*)carve((size_t)kBT * kF * 2);
  __bf16* in1b  = (__bf16*)carve((size_t)kBT * 640 * 2);
  int inD_[4] = {kF, kU, kF + kU, kU};
  __bf16* Wb_[4];
  for (int l = 0; l < 4; ++l) Wb_[l] = (__bf16*)carve((size_t)inD_[l] * k4U * 2);
  __bf16* Wtb_[2];
  for (int b = 0; b < 2; ++b) Wtb_[b] = (__bf16*)carve((size_t)kU * kU * 2);
  __bf16*   hfrag = (__bf16*)carve(4 * 8192 * 2);
  unsigned* ctr   = (unsigned*)carve(16 * 4);

  (void)hipFuncSetAttribute((const void*)lstm_rec_kernel,
                            hipFuncAttributeMaxDynamicSharedMemorySize, (int)kRecLds);

  // ---- weight conversion to bf16
  for (int l = 0; l < 4; ++l)
    convert_bf16_kernel<<<512, 256, 0, stream>>>(W_[l], Wb_[l], (size_t)inD_[l] * k4U);
  for (int b = 0; b < 2; ++b)
    convert_bf16_kernel<<<512, 256, 0, stream>>>(Wt_[b], Wtb_[b], (size_t)kU * kU);
  scatter_x_kernel<<<1024, 256, 0, stream>>>(x, out, xbf, in1b);

  // ---- two highway blocks
  for (int blk = 0; blk < 2; ++blk) {
    const __bf16* inb = (blk == 0) ? xbf : in1b;
    int inD = (blk == 0) ? kF : (kF + kU);
    int l0 = blk * 2, l1 = blk * 2 + 1;
    int colOff = (blk == 0) ? kF : (kF + kU);
    __bf16* nextbf = (blk == 0) ? in1b : nullptr;

    // layer 0: xW GEMM -> recurrence -> LN (fp32 + bf16)
    gemm_bf16_kernel<<<dim3(16, 512), 256, 0, stream>>>(inb, Wb_[l0], b_[l0], xW,
                                                        kBT, k4U, inD, 0);
    init_rec_kernel<<<32, 256, 0, stream>>>(hfrag, ctr);
    lstm_rec_kernel<<<64, 256, kRecLds, stream>>>(xW, U_[l0], hseq, hfrag, ctr);
    layernorm_kernel<<<kBT, 256, 0, stream>>>(hseq, g_[l0], be_[l0], lnA, lnAb);

    // layer 1: xW GEMM -> recurrence -> LN (Hx, fp32 only)
    gemm_bf16_kernel<<<dim3(16, 512), 256, 0, stream>>>(lnAb, Wb_[l1], b_[l1], xW,
                                                        kBT, k4U, kU, 0);
    init_rec_kernel<<<32, 256, 0, stream>>>(hfrag, ctr);
    lstm_rec_kernel<<<64, 256, kRecLds, stream>>>(xW, U_[l1], hseq, hfrag, ctr);
    layernorm_kernel<<<kBT, 256, 0, stream>>>(hseq, g_[l1], be_[l1], lnB, nullptr);

    // transform gate: Tx = sigmoid(out @ Wt + bt), then highway merge
    gemm_bf16_kernel<<<dim3(4, 512), 256, 0, stream>>>(lnAb, Wtb_[blk], bt_[blk], TxB,
                                                       kBT, kU, kU, 1);
    highway_kernel<<<4096, 256, 0, stream>>>(lnB, TxB, lnA, out, colOff, nextbf);
  }
}
